// Conv2DMod_48515950575806
// MI455X (gfx1250) — matrix operand order repaired and checked by reading, not compiled, not run on hardware
//
#include <hip/hip_runtime.h>
#include <hip/hip_bf16.h>

typedef __attribute__((ext_vector_type(16))) __bf16 v16bf;
typedef __attribute__((ext_vector_type(8)))  float  v8f;

#define BATCH 8
#define CIN   64
#define COUT  64
#define HDIM  256
#define WDIM  256
#define HW    (HDIM * WDIM)

// ---------------------------------------------------------------------------
// Kernel 1: modulate + demodulate weights, pack bf16 into WMMA A-fragment
// layout.  One wave per (b, cout).  K-order: k = tap*64 + cin, chunked by 32.
// A 16x32 bf16 fragment layout as packed here (contiguous k, same as B side):
//   lane L (m = L%16), slot s: k = s + (L>=16 ? 16:0)
// Packed block per (b, mt, tap, cc): 32 lanes x 16 bf16 contiguous.
// ---------------------------------------------------------------------------
__global__ __launch_bounds__(256) void modpack_kernel(
    const float* __restrict__ s, const float* __restrict__ weight,
    __bf16* __restrict__ apack)
{
  int tid  = threadIdx.x;
  int lane = tid & 31;
  int gw   = blockIdx.x * 8 + (tid >> 5);   // 512 waves total
  int b    = gw >> 6;
  int cout = gw & 63;

  const float* wrow = weight + cout * (CIN * 9);
  const float* srow = s + b * CIN;

  float v[18];
  float sum = 0.f;
#pragma unroll
  for (int i = 0; i < 18; ++i) {
    int e   = i * 32 + lane;          // e = cin*9 + tap, coalesced
    int cin = e / 9;
    float val = wrow[e] * (srow[cin] + 1.0f);
    v[i] = val;
    sum += val * val;
  }
#pragma unroll
  for (int off = 16; off > 0; off >>= 1)
    sum += __shfl_xor(sum, off, 32);
  float d = rsqrtf(sum + 1e-8f);

  int mt = cout >> 4;
  int m  = cout & 15;
#pragma unroll
  for (int i = 0; i < 18; ++i) {
    int e    = i * 32 + lane;
    int cin  = e / 9;
    int tap  = e - cin * 9;
    int cc   = cin >> 5;              // 32-wide K chunk
    int kk   = cin & 31;
    int hi   = (kk >> 4) & 1;         // lanes 16-31 hold kk in 16-31
    int p    = ((kk & 16) ? 4 : 0) + ((kk & 7) >> 1);
    int slot = kk & 15;
    int lp   = m + hi * 16;
    size_t idx = ((((size_t)(b * 4 + mt) * 9 + tap) * 2 + cc) * 32 + lp) * 16 + slot;
    apack[idx] = (__bf16)(v[i] * d);
  }
}

// ---------------------------------------------------------------------------
// Main kernel: implicit-GEMM conv via v_wmma_f32_16x16x32_bf16.
// Workgroup = 256 threads (8 waves), output tile 4 rows x 64 cols, all 64 couts.
// LDS: input halo tile 6 rows x 66 cols x 64 cin, bf16, [pixel][cin] layout,
// pixel stride 72 (144 B: 16B-aligned b128 loads, spread across banks).
// Wave w: cout block mt = w&3, tile rows 2*(w>>2)..+1  -> 8 C tiles (64 VGPRs).
// K loop fully unrolled (9 taps x 2 cin-chunks): all LDS/global offsets become
// immediates off a single lane base; B fragments for a k-step are loaded into
// a dedicated 8-fragment array BEFORE the WMMAs so the scheduler can clause
// the 16 ds_load_b128 and drain with staggered dscnt waits instead of
// s_wait_dscnt 0 per WMMA.
// ---------------------------------------------------------------------------
#define LROWS 6
#define LCOLS 66
#define LPS   72   // padded pixel stride in bf16 elems (144 B, 16B-aligned)

__global__ __launch_bounds__(256) void conv_kernel(
    const float* __restrict__ x, const __bf16* __restrict__ apack,
    float* __restrict__ out)
{
  __shared__ __align__(16) __bf16 xs[LROWS * LCOLS * LPS];   // 57024 B

  int tid  = threadIdx.x;
  int lane = tid & 31;
  int wave = tid >> 5;

  int bid = blockIdx.x;
  int wt  = bid & 3;
  int ht  = (bid >> 2) & 63;
  int b   = bid >> 8;
  int w0  = wt * 64;
  int h0  = ht * 4;

  // ---- stage x tile (f32 -> bf16) into LDS, coalesced along columns
  const float* xb = x + (size_t)b * CIN * HW;
  for (int i = tid; i < CIN * LROWS * LCOLS; i += 256) {
    int cin = i / (LROWS * LCOLS);
    int rem = i - cin * (LROWS * LCOLS);
    int row = rem / LCOLS;
    int col = rem - row * LCOLS;
    int gr = h0 - 1 + row;
    int gc = w0 - 1 + col;
    float val = 0.f;
    if ((unsigned)gr < HDIM && (unsigned)gc < WDIM)
      val = xb[(size_t)cin * HW + gr * WDIM + gc];
    xs[(row * LCOLS + col) * LPS + cin] = (__bf16)val;
  }
  __syncthreads();

  int mt    = wave & 3;
  int rbase = (wave >> 2) * 2;
  int ncol  = lane & 15;
  int chalf = (lane >> 4) * 16;        // B layout: lanes 16-31 hold K=16..31

  v8f zero = {};
  v8f acc[8];
#pragma unroll
  for (int i = 0; i < 8; ++i) acc[i] = zero;

  // single lane-dependent bases; everything else becomes immediate offsets
  const __bf16* abase = apack + (size_t)(b * 4 + mt) * (9 * 2 * 512) + lane * 16;
  const __bf16* xbase = &xs[(rbase * LCOLS + ncol) * LPS + chalf];

#pragma unroll
  for (int t = 0; t < 9; ++t) {        // tap = kh*3 + kw  (compile-time)
    const int kh = t / 3;
    const int kw = t - kh * 3;
#pragma unroll
    for (int cc = 0; cc < 2; ++cc) {
      // A fragment: 32 B/lane contiguous, L2-resident
      v16bf afrag = *(const v16bf*)(abase + (t * 2 + cc) * 512);
      // load ALL 8 B fragments first -> one ds clause, overlapped waits
      v16bf bf[8];
#pragma unroll
      for (int ct = 0; ct < 8; ++ct) {
        const int off = (((ct >> 2) + kh) * LCOLS + (ct & 3) * 16 + kw) * LPS + cc * 32;
        bf[ct] = *(const v16bf*)(xbase + off);
      }
#pragma unroll
      for (int ct = 0; ct < 8; ++ct) {
        acc[ct] = __builtin_amdgcn_wmma_f32_16x16x32_bf16(
            false, afrag, false, bf[ct], (short)0, acc[ct], false, false);
      }
    }
  }

  // ---- epilogue: C 16x16 f32 layout -> out[b][cout][h][w]
  float* ob = out + (size_t)b * COUT * HW;
  int coutbase = mt * 16 + (lane >> 4) * 8;   // lanes 16-31 hold M = r+8
#pragma unroll
  for (int ct = 0; ct < 8; ++ct) {
    int trow = rbase + (ct >> 2);
    int cb   = ct & 3;
    int gr = h0 + trow;
    int gc = w0 + cb * 16 + ncol;
#pragma unroll
    for (int r = 0; r < 8; ++r) {
      ob[(size_t)(coutbase + r) * HW + gr * WDIM + gc] = acc[ct][r];
    }
  }
}

extern "C" void kernel_launch(void* const* d_in, const int* in_sizes, int n_in,
                              void* d_out, int out_size, void* d_ws, size_t ws_size,
                              hipStream_t stream) {
  const float* x = (const float*)d_in[0];
  const float* s = (const float*)d_in[1];
  const float* w = (const float*)d_in[2];
  float* out = (float*)d_out;
  __bf16* apack = (__bf16*)d_ws;      // 8*4*9*2*512 bf16 = 576 KB

  modpack_kernel<<<64, 256, 0, stream>>>(s, w, apack);
  conv_kernel<<<BATCH * (HDIM / 4) * (WDIM / 64), 256, 0, stream>>>(x, apack, out);
}
